// SmoothQuantW8A16Linear_12008728559686
// MI455X (gfx1250) — matrix operand order repaired and checked by reading, not compiled, not run on hardware
//
#include <hip/hip_runtime.h>
#include <stdint.h>

// ---------------------------------------------------------------------------
// SmoothQuant W8A16 linear as a bf16 WMMA GEMM for gfx1250 (MI455X, wave32).
//   out[M,N] = (x * smooth_scale) @ (w_int8 * weight_scale)^T + bias
// M = 4*2048 = 8192, N = 4096, K = 4096.
// Scales are per-tensor -> folded into the fp32 epilogue.
// ---------------------------------------------------------------------------

typedef __attribute__((ext_vector_type(16))) __bf16 v16bf;
typedef __attribute__((ext_vector_type(8)))  __bf16 v8bf;
typedef __attribute__((ext_vector_type(4)))  __bf16 v4bf;
typedef __attribute__((ext_vector_type(8)))  float  v8f;

#define MDIM 8192
#define NDIM 4096
#define KDIM 4096

#define BM 128          // block tile M
#define BN 128          // block tile N
#define BK 64           // K step staged in LDS
#define LDSK 72         // padded K stride (elements) -> 144B rows, 16B aligned
#define MT 2            // 16x16 tiles per wave along M (32 rows)
#define NT 4            // 16x16 tiles per wave along N (64 cols)

__global__ __launch_bounds__(256) void sq_w8a16_wmma_kernel(
    const float*  __restrict__ x,      // [M, K] fp32
    const int8_t* __restrict__ w,      // [N, K] int8
    const float*  __restrict__ wscale, // [1]
    const float*  __restrict__ sscale, // [1]
    const float*  __restrict__ bias,   // [N]
    float*        __restrict__ out)    // [M, N]
{
    __shared__ __bf16 As[2][BM * LDSK];  // 36 KB
    __shared__ __bf16 Bs[2][BN * LDSK];  // 36 KB

    const int tid  = threadIdx.x;
    const int lane = tid & 31;
    const int wave = tid >> 5;
    const int wm   = (wave & 3) * (MT * 16);  // 4 waves along M -> 0,32,64,96
    const int wn   = (wave >> 2) * (NT * 16); // 2 waves along N -> 0,64
    const int half = lane >> 4;               // 0: lanes 0-15, 1: lanes 16-31
    const int l16  = lane & 15;

    const int bm = blockIdx.y * BM;
    const int bn = blockIdx.x * BN;

    v8f acc[MT][NT];
#pragma unroll
    for (int mt = 0; mt < MT; ++mt)
#pragma unroll
        for (int nt = 0; nt < NT; ++nt)
            acc[mt][nt] = (v8f){0.f, 0.f, 0.f, 0.f, 0.f, 0.f, 0.f, 0.f};

    // ---- stage one BMxBK (fp32->bf16) and BNxBK (int8->bf16) tile into LDS
    auto loadTile = [&](int t, int buf) {
        const int kb = t * BK;
        // A: 128 rows x 64 fp32 = 2048 float4 chunks, 8 per thread
#pragma unroll
        for (int i = 0; i < 8; ++i) {
            const int idx = tid + i * 256;
            const int row = idx >> 4;          // 0..127
            const int c4  = (idx & 15) * 4;    // 0..60
            const float4 v =
                *(const float4*)(x + (size_t)(bm + row) * KDIM + kb + c4);
            v4bf p;
            p.x = (__bf16)v.x; p.y = (__bf16)v.y;
            p.z = (__bf16)v.z; p.w = (__bf16)v.w;
            *(v4bf*)&As[buf][row * LDSK + c4] = p;
        }
        // B: 128 rows x 64 int8 = 512 16-byte chunks, 2 per thread
#pragma unroll
        for (int i = 0; i < 2; ++i) {
            const int idx  = tid + i * 256;
            const int row  = idx >> 2;         // 0..127 (output feature n)
            const int c16  = (idx & 3) * 16;   // 0,16,32,48
            union { int4 v; int8_t b[16]; } u;
            u.v = *(const int4*)(w + (size_t)(bn + row) * KDIM + kb + c16);
            v8bf lo, hi;
#pragma unroll
            for (int j = 0; j < 8; ++j) {
                lo[j] = (__bf16)(float)u.b[j];
                hi[j] = (__bf16)(float)u.b[j + 8];
            }
            *(v8bf*)&Bs[buf][row * LDSK + c16]     = lo;
            *(v8bf*)&Bs[buf][row * LDSK + c16 + 8] = hi;
        }
        // prefetch next weight tile into cache hierarchy (global_prefetch_b8)
        if (kb + BK < KDIM) {
            __builtin_prefetch(
                w + (size_t)(bn + (tid >> 1)) * KDIM + (kb + BK) + (tid & 1) * 32,
                0, 1);
        }
    };

    // ---- one staged tile: 2 k-steps of v_wmma_f32_16x16x32_bf16
    auto mmaTile = [&](int buf) {
#pragma unroll
        for (int ks = 0; ks < BK / 32; ++ks) {
            v16bf afrag[MT];
            v16bf bfrag[NT];
#pragma unroll
            for (int mt = 0; mt < MT; ++mt) {
                // A 16x32 bf16 layout: lane<16 -> K {0..7, 16..23},
                //                      lane>=16 -> K {8..15, 24..31}
                const __bf16* ap =
                    &As[buf][(wm + mt * 16 + l16) * LDSK + ks * 32 + half * 8];
                v8bf lo = *(const v8bf*)ap;          // 16B ds_load
                v8bf hi = *(const v8bf*)(ap + 16);   // +32B
                afrag[mt] = __builtin_shufflevector(
                    lo, hi, 0, 1, 2, 3, 4, 5, 6, 7, 8, 9, 10, 11, 12, 13, 14, 15);
            }
#pragma unroll
            for (int nt = 0; nt < NT; ++nt) {
                // B 32x16 bf16 layout: lane<16 -> K {0..7, 16..23},
                //                      lane>=16 -> K {8..15, 24..31}
                const __bf16* bp =
                    &Bs[buf][(wn + nt * 16 + l16) * LDSK + ks * 32 + half * 8];
                v8bf lo = *(const v8bf*)bp;
                v8bf hi = *(const v8bf*)(bp + 16);
                bfrag[nt] = __builtin_shufflevector(
                    lo, hi, 0, 1, 2, 3, 4, 5, 6, 7, 8, 9, 10, 11, 12, 13, 14, 15);
            }
#pragma unroll
            for (int mt = 0; mt < MT; ++mt)
#pragma unroll
                for (int nt = 0; nt < NT; ++nt)
                    acc[mt][nt] = __builtin_amdgcn_wmma_f32_16x16x32_bf16(
                        /*neg_a=*/false, afrag[mt],
                        /*neg_b=*/false, bfrag[nt],
                        /*c_mod=*/(short)0, acc[mt][nt],
                        /*reuse_a=*/false, /*reuse_b=*/false);
        }
    };

    // ---- main loop: double-buffered K sweep
    loadTile(0, 0);
    __syncthreads();
    const int nT = KDIM / BK;  // 64
    for (int t = 0; t < nT; ++t) {
        const int cur = t & 1;
        if (t + 1 < nT) loadTile(t + 1, cur ^ 1);
        mmaTile(cur);
        __syncthreads();
    }

    // ---- epilogue: fold per-tensor scales + bias, fp32 store
    const float scale = wscale[0] * sscale[0];
#pragma unroll
    for (int nt = 0; nt < NT; ++nt) {
        const int n  = bn + wn + nt * 16 + l16;
        const float bv = bias[n];
#pragma unroll
        for (int mt = 0; mt < MT; ++mt) {
#pragma unroll
            for (int r = 0; r < 8; ++r) {
                // C/D layout: VGPR r -> M = r (+8 for lanes 16-31), N = lane%16
                const int m = bm + wm + mt * 16 + half * 8 + r;
                out[(size_t)m * NDIM + n] = acc[mt][nt][r] * scale + bv;
            }
        }
    }
}

extern "C" void kernel_launch(void* const* d_in, const int* in_sizes, int n_in,
                              void* d_out, int out_size, void* d_ws, size_t ws_size,
                              hipStream_t stream) {
    (void)in_sizes; (void)n_in; (void)out_size; (void)d_ws; (void)ws_size;
    const float*  x      = (const float*)d_in[0];
    const int8_t* w8     = (const int8_t*)d_in[1];
    const float*  wscale = (const float*)d_in[2];
    const float*  sscale = (const float*)d_in[3];
    const float*  bias   = (const float*)d_in[4];
    float*        out    = (float*)d_out;

    dim3 grid(NDIM / BN, MDIM / BM);  // (32, 64)
    dim3 block(256);                  // 8 wave32 waves
    sq_w8a16_wmma_kernel<<<grid, block, 0, stream>>>(x, w8, wscale, sscale,
                                                     bias, out);
}
